// ProjectionAggregating_85839216378535
// MI455X (gfx1250) — matrix operand order repaired and checked by reading, not compiled, not run on hardware
//
#include <hip/hip_runtime.h>

// ---------------------------------------------------------------------------
// ProjectionAggregating on MI455X (gfx1250, wave32, WMMA bf16 16x16x32)
//   B=4096, N=64, D=128, K1=256
// One workgroup (256 threads = 8 waves) per batch row b.
//   wave w owns output dims d in [16w, 16w+16); 4 M-tiles of 16 neighbors.
// Round 2: replace manual bf16 bit-twiddling with native __bf16 /
// __builtin_convertvector (packed cvt), vectorize fill x4, and replace the
// serial 128-iter L1-norm loops with shfl_xor tree reductions.
// ---------------------------------------------------------------------------

typedef __attribute__((ext_vector_type(16))) __bf16 v16bf;
typedef __attribute__((ext_vector_type(4)))  __bf16 v4bf;
typedef __attribute__((ext_vector_type(8)))  float  v8f;
typedef __attribute__((ext_vector_type(4)))  float  v4f;

#define D_DIM   128
#define N_NBR   64
#define K1      256
#define RR_STRIDE 136   // bf16 per row (128 + 8 pad) -> 272B, bank-conflict free
#define H1_STRIDE 136

// LDS layout (dynamic), all 16B-aligned offsets:
#define OFF_BIAS 0                       // 64*128 f32            = 32768
#define OFF_FR   (OFF_BIAS + 32768)      // 128 f32               = 512
#define OFF_QE   (OFF_FR + 512)          // 128 f32               = 512
#define OFF_RR   (OFF_QE + 512)          // 64*136 bf16           = 17408
#define OFF_H1   (OFF_RR + 17408)        // 64*136 bf16           = 17408
#define OFF_QR   (OFF_H1 + 17408)        // 128 bf16 (pad 256B)   = 256
#define OFF_WS   (OFF_QR + 256)          // 8 f32 + 4 f32 (pad 64)= 64
#define SMEM_BYTES (OFF_WS + 64)         // 68928 B (<320KB/WGP; ~4 WGs resident)

union Frag16 {
    uint4 q[2];
    v4bf  b4[4];
    v16bf v;
};

// B-fragment (K x 16, bf16): lane holds 16 consecutive K values of column
// d = 16w + (lane&15), K base = k0 + (lane>>4)*16.  W row-major [d,k] gives
// 16 consecutive floats -> 64B load + packed cvt.
__device__ __forceinline__ v16bf load_b_frag(const float* __restrict__ wrow16) {
    const v4f* p = (const v4f*)wrow16;
    Frag16 f;
    f.b4[0] = __builtin_convertvector(p[0], v4bf);
    f.b4[1] = __builtin_convertvector(p[1], v4bf);
    f.b4[2] = __builtin_convertvector(p[2], v4bf);
    f.b4[3] = __builtin_convertvector(p[3], v4bf);
    return f.v;
}

// A-fragment (16 x 32, bf16): lane needs K = ka..ka+15 (VGPR0-7),
// ka = k0 + (lane>>4)*16 -> two b128 LDS loads
// 16 bytes apart.
__device__ __forceinline__ v16bf load_a_frag(const __bf16* __restrict__ base) {
    Frag16 f;
    f.q[0] = *(const uint4*)(base);
    f.q[1] = *(const uint4*)(base + 8);
    return f.v;
}

__global__ __launch_bounds__(256)
void ProjectionAggregating_85839216378535_kernel(
        const float* __restrict__ query_emb,   // [B,128]
        const float* __restrict__ refer_embs,  // [B,64,128]
        const float* __restrict__ query_r,     // [B,128]
        const float* __restrict__ refer_r,     // [B,64,128]
        const float* __restrict__ start_embs,  // [B,64,128]
        const float* __restrict__ W1,          // [128,256] (d,k)
        const float* __restrict__ b1,          // [128]
        const float* __restrict__ W2,          // [128,128] (d,k)
        const float* __restrict__ b2,          // [128]
        float* __restrict__ out)               // [B,128]
{
    extern __shared__ char smem[];
    float*  bias_lds = (float*)(smem + OFF_BIAS);
    float*  fr_lds   = (float*)(smem + OFF_FR);
    float*  qe_lds   = (float*)(smem + OFF_QE);
    __bf16* rr_lds   = (__bf16*)(smem + OFF_RR);
    __bf16* h1_lds   = (__bf16*)(smem + OFF_H1);
    __bf16* qr_lds   = (__bf16*)(smem + OFF_QR);
    float*  ws_lds   = (float*)(smem + OFF_WS);      // [0..7]  per-wave |fr| sums
    float*  qs_lds   = (float*)(smem + OFF_WS + 32); // [0..3]  per-wave |qe| sums

    const int b    = blockIdx.x;
    const int tid  = threadIdx.x;
    const int wv   = tid >> 5;          // wave id = ntile (0..7)
    const int lane = tid & 31;
    const size_t big = (size_t)b * (N_NBR * D_DIM);

    const float* rr = refer_r    + big;
    const float* re = refer_embs + big;
    const float* se = start_embs + big;

    // ---- Fill phase: stream the three big tensors ONCE from HBM ----------
    // (vectorized x4: three global_load_b128 per iter, bias b128 store,
    //  refer_r packed-cvt + b64 store)
    for (int i = tid * 4; i < N_NBR * D_DIM; i += 256 * 4) {
        int n = i >> 7, k = i & 127;
        v4f rv  = *(const v4f*)(rr + i);
        v4f rev = *(const v4f*)(re + i);
        v4f sev = *(const v4f*)(se + i);
        v4f bia = rev - sev * rv;
        *(v4f*)(bias_lds + i) = bia;
        *(v4bf*)(rr_lds + n * RR_STRIDE + k) = __builtin_convertvector(rv, v4bf);
    }
    if (tid < D_DIM) {                  // waves 0..3 fully active (uniform)
        float q = query_emb[(size_t)b * D_DIM + tid];
        qe_lds[tid] = q;
        qr_lds[tid] = (__bf16)query_r[(size_t)b * D_DIM + tid];
        // query L1 norm: full-wave tree reduce, 4 wave partials
        float s = fabsf(q);
        s += __shfl_xor(s, 1, 32);
        s += __shfl_xor(s, 2, 32);
        s += __shfl_xor(s, 4, 32);
        s += __shfl_xor(s, 8, 32);
        s += __shfl_xor(s, 16, 32);
        if (lane == 0) qs_lds[wv] = s;
    }
    __syncthreads();

    const int dcol  = (wv << 4) + (lane & 15);   // output dim of this lane
    const int khalf = lane >> 4;                 // 0 or 1 (this code's order: lane half h holds k = 16h..16h+15)

    // ====================== GEMM1: h1 = relu(cat . W1^T + b1) =============
    // K in [0,128)  : cat = query_r (identical rows) -> ONE accumulator tile
    // K in [128,256): cat = refer_r                  -> 4 M-tiles
    const float* w1row = W1 + (size_t)dcol * K1;

    v8f accq = {};
    #pragma unroll
    for (int k0 = 0; k0 < 128; k0 += 32) {
        v16bf bf = load_b_frag(w1row + k0 + (khalf << 4));
        v16bf af = load_a_frag(qr_lds + k0 + (khalf << 4));
        accq = __builtin_amdgcn_wmma_f32_16x16x32_bf16(
                   false, af, false, bf, (short)0, accq, false, false);
    }

    v8f acc[4] = {};
    #pragma unroll
    for (int k0 = 0; k0 < 128; k0 += 32) {
        v16bf bf = load_b_frag(w1row + 128 + k0 + (khalf << 4));
        const int ka = k0 + (khalf << 4);
        #pragma unroll
        for (int m = 0; m < 4; ++m) {
            int row = (m << 4) + (lane & 15);
            v16bf af = load_a_frag(rr_lds + row * RR_STRIDE + ka);
            acc[m] = __builtin_amdgcn_wmma_f32_16x16x32_bf16(
                         false, af, false, bf, (short)0, acc[m], false, false);
        }
    }

    // epilogue: +b1, +query-part, relu, store h1 as bf16 (A-source of GEMM2)
    {
        float b1v = b1[dcol];
        #pragma unroll
        for (int m = 0; m < 4; ++m) {
            int rowbase = (m << 4) + (khalf << 3);   // C layout: M = r + 8*khalf
            #pragma unroll
            for (int r = 0; r < 8; ++r) {
                float h = acc[m][r] + accq[r] + b1v;
                h = h > 0.0f ? h : 0.0f;
                h1_lds[(rowbase + r) * H1_STRIDE + dcol] = (__bf16)h;
            }
        }
    }
    __syncthreads();

    // ====================== GEMM2: attn = h1 . W2^T + b2 ==================
    const float* w2row = W2 + (size_t)dcol * D_DIM;
    v8f acc2[4] = {};
    #pragma unroll
    for (int k0 = 0; k0 < 128; k0 += 32) {
        v16bf bf = load_b_frag(w2row + k0 + (khalf << 4));
        const int ka = k0 + (khalf << 4);
        #pragma unroll
        for (int m = 0; m < 4; ++m) {
            int row = (m << 4) + (lane & 15);
            v16bf af = load_a_frag(h1_lds + row * H1_STRIDE + ka);
            acc2[m] = __builtin_amdgcn_wmma_f32_16x16x32_bf16(
                          false, af, false, bf, (short)0, acc2[m], false, false);
        }
    }

    // ---- fused epilogue: (attn+b2)*bias, reduce over the 64 neighbors ----
    float b2v = b2[dcol];
    float partial = 0.0f;
    #pragma unroll
    for (int m = 0; m < 4; ++m) {
        int rowbase = (m << 4) + (khalf << 3);
        #pragma unroll
        for (int r = 0; r < 8; ++r) {
            float bia = bias_lds[(rowbase + r) * D_DIM + dcol];
            partial = fmaf(acc2[m][r] + b2v, bia, partial);
        }
    }
    // lanes l and l+16 hold the two disjoint halves of the 64 neighbors;
    // after this, fr is identical in lanes l and l+16.
    float fr = partial + __shfl_xor(partial, 16, 32);
    if (lane < 16) fr_lds[dcol] = fr;

    // wave's contribution to ref L1 norm: sum |fr| over the 16 distinct d's
    // (each 16-lane half holds the same 16 values -> xor-tree over 1,2,4,8)
    {
        float s = fabsf(fr);
        s += __shfl_xor(s, 1, 32);
        s += __shfl_xor(s, 2, 32);
        s += __shfl_xor(s, 4, 32);
        s += __shfl_xor(s, 8, 32);
        if (lane == 0) ws_lds[wv] = s;
    }
    __syncthreads();

    // ---- L1 norms + scaled residual --------------------------------------
    if (tid < D_DIM) {
        float rn = ((ws_lds[0] + ws_lds[1]) + (ws_lds[2] + ws_lds[3]))
                 + ((ws_lds[4] + ws_lds[5]) + (ws_lds[6] + ws_lds[7]));
        float qn = (qs_lds[0] + qs_lds[1]) + (qs_lds[2] + qs_lds[3]);
        float scale = 1e-9f + (rn / qn) * 2.5f;
        out[(size_t)b * D_DIM + tid] = qe_lds[tid] + fr_lds[tid] / scale;
    }
}

extern "C" void kernel_launch(void* const* d_in, const int* in_sizes, int n_in,
                              void* d_out, int out_size, void* d_ws, size_t ws_size,
                              hipStream_t stream) {
    const float* query_emb  = (const float*)d_in[0];
    const float* refer_embs = (const float*)d_in[1];
    const float* query_r    = (const float*)d_in[2];
    const float* refer_r    = (const float*)d_in[3];
    const float* start_embs = (const float*)d_in[4];
    const float* W1         = (const float*)d_in[5];
    const float* b1         = (const float*)d_in[6];
    const float* W2         = (const float*)d_in[7];
    const float* b2         = (const float*)d_in[8];
    float* out = (float*)d_out;

    int Bv = in_sizes[0] / D_DIM;   // 4096
    hipLaunchKernelGGL(ProjectionAggregating_85839216378535_kernel,
                       dim3(Bv), dim3(256), SMEM_BYTES, stream,
                       query_emb, refer_embs, query_r, refer_r, start_embs,
                       W1, b1, W2, b2, out);
}
